// LeanAlbertLayer_73942156967955
// MI455X (gfx1250) — matrix operand order repaired and checked by reading, not compiled, not run on hardware
//
#include <hip/hip_runtime.h>
#include <hip/hip_bf16.h>

// ---------------------------------------------------------------------------
// LeanAlbertLayer forward for MI455X (gfx1250), bf16 WMMA path.
// B=4, S=1024, H=1024, NH=16, HD=64, I=4096.
// ---------------------------------------------------------------------------

#define BB   4
#define SS   1024
#define HH   1024
#define NHD  16
#define HDD  64
#define II   4096
#define TT   (BB * SS)      // 4096 tokens
#define H3   (3 * HH)       // 3072
#define I2   (2 * II)       // 8192

typedef __attribute__((ext_vector_type(16))) __bf16 bf16x16;
typedef __attribute__((ext_vector_type(8)))  float  f32x8;

union Frag {
    uint4   u4[2];
    uint2   u2[4];
    bf16x16 v;
};

// ---- WMMA wrapper -----------------------------------------------------------
__device__ __forceinline__ f32x8 wmma_bf16(bf16x16 a, bf16x16 b, f32x8 c) {
    return __builtin_amdgcn_wmma_f32_16x16x32_bf16(
        /*neg_a=*/false, a, /*neg_b=*/false, b,
        /*c_mod=*/(short)0, c, /*reuse_a=*/false, /*reuse_b=*/false);
}

// ---- CDNA5 async copy: global -> LDS (ASYNCcnt-tracked, ISA 15.18.3) -------
__device__ __forceinline__ void async_load_b128(const void* gptr, unsigned ldsOff) {
    asm volatile("global_load_async_to_lds_b128 %0, %1, off"
                 :: "v"(ldsOff), "v"((unsigned long long)(size_t)gptr)
                 : "memory");
}
__device__ __forceinline__ void wait_asynccnt0() {
    asm volatile("s_wait_asynccnt 0" ::: "memory");
}

// ---- Fragment loaders (ISA 7.12.2 layouts) ---------------------------------
// A-matrix 16x32 bf16: lane L (m=L&15, half=L>>4) holds
//   a[0..7]  = row m, K = half*8 + 0..7
//   a[8..15] = row m, K = 16 + half*8 + 0..7
__device__ __forceinline__ bf16x16 load_a_frag_lds(const __bf16* base, int stride, int lane) {
    int m    = lane & 15;
    int half = (lane >> 4) & 1;
    const __bf16* row = base + m * stride + half * 8;
    Frag f;
    f.u2[0] = *(const uint2*)(row);
    f.u2[1] = *(const uint2*)(row + 4);
    f.u2[2] = *(const uint2*)(row + 16);
    f.u2[3] = *(const uint2*)(row + 20);
    return f.v;
}

__device__ __forceinline__ bf16x16 load_a_frag_global(const __bf16* A, long ld, int m0, int k0, int lane) {
    int m    = m0 + (lane & 15);
    int half = (lane >> 4) & 1;
    const __bf16* p = A + (long)m * ld + k0 + half * 8;
    Frag f;
    f.u4[0] = *(const uint4*)(p);
    f.u4[1] = *(const uint4*)(p + 16);
    return f.v;
}

// B-matrix 32x16 bf16: lane L (n=L&15, koff=(L>>4)*8) holds b[0..7] = B[koff+0..7, n], b[8..15] = B[16+koff+0..7, n].
// Source is row-major [N, K] (B[k,n] = W[n,k]) -> two runs of 8 contiguous bf16 per lane.
__device__ __forceinline__ bf16x16 load_b_frag(const __bf16* W, long ld, int n0, int k0, int lane) {
    int n    = n0 + (lane & 15);
    int koff = ((lane >> 4) & 1) * 8;
    const __bf16* p = W + (long)n * ld + k0 + koff;
    Frag f;
    f.u4[0] = *(const uint4*)(p);
    f.u4[1] = *(const uint4*)(p + 16);
    return f.v;
}

// ---------------------------------------------------------------------------
// f32 -> bf16 weight conversion
// ---------------------------------------------------------------------------
__global__ __launch_bounds__(256) void cvt_bf16_kernel(const float* __restrict__ src,
                                                       __bf16* __restrict__ dst, long n) {
    long g = (long)blockIdx.x * 256 + threadIdx.x;
    if (g < n) dst[g] = (__bf16)src[g];
}

// ---------------------------------------------------------------------------
// LayerNorm: one block (128 threads) per token, f32 in -> bf16 out
// ---------------------------------------------------------------------------
__global__ __launch_bounds__(128) void ln_kernel(const float* __restrict__ x,
                                                 const float* __restrict__ w,
                                                 const float* __restrict__ b,
                                                 __bf16* __restrict__ y) {
    __shared__ float red[8];
    long t = blockIdx.x;
    const float* xr = x + t * HH;
    float s = 0.f, s2 = 0.f;
    for (int i = threadIdx.x; i < HH; i += 128) {
        float v = xr[i];
        s += v; s2 += v * v;
    }
    #pragma unroll
    for (int off = 16; off >= 1; off >>= 1) {
        s  += __shfl_xor(s, off, 32);
        s2 += __shfl_xor(s2, off, 32);
    }
    int wave = threadIdx.x >> 5, lane = threadIdx.x & 31;
    if (lane == 0) { red[wave] = s; red[4 + wave] = s2; }
    __syncthreads();
    if (threadIdx.x == 0) {
        red[0] = red[0] + red[1] + red[2] + red[3];
        red[4] = red[4] + red[5] + red[6] + red[7];
    }
    __syncthreads();
    float mu  = red[0] * (1.0f / HH);
    float var = red[4] * (1.0f / HH) - mu * mu;
    float inv = rsqrtf(var + 1e-12f);
    for (int i = threadIdx.x; i < HH; i += 128)
        y[t * HH + i] = (__bf16)((xr[i] - mu) * inv * w[i] + b[i]);
}

// ---------------------------------------------------------------------------
// Tiled bf16 GEMM: out[m,n] = sum_k A[m,k] * W[n,k]  (+bias, +resid)
// Block = 256 threads = 8 waves (2 x 4), block tile 128(M) x 128(N),
// wave tile 64x32 = 4x2 WMMA accumulators (8 WMMA per 32-K slab).
// A tile: double-buffered LDS filled with global_load_async_to_lds_b128
// (ASYNCcnt), copy of slab k+1 overlaps WMMAs on slab k; ONE barrier/slab.
// W streamed from global (bf16 weights total 32 MB -> L2-resident).
// ---------------------------------------------------------------------------
__global__ __launch_bounds__(256) void gemm_bf16_kernel(
    const __bf16* __restrict__ A, const __bf16* __restrict__ W,
    const float* __restrict__ bias, const float* __restrict__ resid,
    float* __restrict__ outF, __bf16* __restrict__ outB,
    int M, int N, int K)
{
    __shared__ __bf16 As[2][128][32];   // 2 x 8 KB ping-pong, 64 B rows (16B-aligned)

    const int tid  = threadIdx.x;
    const int lane = tid & 31;
    const int wave = tid >> 5;
    const int wm = wave >> 2;            // 0..1  -> 64-row strip
    const int wn = wave & 3;             // 0..3  -> 32-col strip
    const int m0 = blockIdx.y * 128;
    const int n0 = blockIdx.x * 128;
    const int wN = n0 + wn * 32;

    // async A-tile copy assignment: 2 threads per row, 16 bf16 (2 x b128) each
    const int arow = tid >> 1;           // 0..127
    const int acol = (tid & 1) * 16;     // 0 or 16
    const __bf16* Ab = A + (long)(m0 + arow) * K + acol;
    const unsigned lds0 = (unsigned)(size_t)&As[0][arow][acol];
    const unsigned lds1 = (unsigned)(size_t)&As[1][arow][acol];

    f32x8 acc[4][2] = {};

    // prologue: slab 0 -> buffer 0
    async_load_b128(Ab,     lds0);
    async_load_b128(Ab + 8, lds0 + 16);
    wait_asynccnt0();
    __syncthreads();

    int buf = 0;
    for (int kb = 0; kb < K; kb += 32) {
        if (kb + 32 < K) {
            const __bf16* g = Ab + kb + 32;
            unsigned dst = buf ? lds0 : lds1;
            async_load_b128(g,     dst);
            async_load_b128(g + 8, dst + 16);
            __builtin_prefetch(W + (long)(wN + (lane & 15)) * K + kb + 64, 0, 1);
        }

        bf16x16 b0 = load_b_frag(W, K, wN,      kb, lane);
        bf16x16 b1 = load_b_frag(W, K, wN + 16, kb, lane);
        #pragma unroll
        for (int i = 0; i < 4; ++i) {
            bf16x16 a = load_a_frag_lds(&As[buf][wm * 64 + i * 16][0], 32, lane);
            acc[i][0] = wmma_bf16(a, b0, acc[i][0]);
            acc[i][1] = wmma_bf16(a, b1, acc[i][1]);
        }

        wait_asynccnt0();    // our asyncs for next slab are done
        __syncthreads();     // everyone's are done; prev buffer free for reuse
        buf ^= 1;
    }

    // Epilogue: C layout -> lane n = lane&15, VGPR r = row r + (lane>=16)*8
    const int half = (lane >> 4) & 1;
    const int nc   = lane & 15;
    #pragma unroll
    for (int i = 0; i < 4; ++i) {
        #pragma unroll
        for (int j = 0; j < 2; ++j) {
            #pragma unroll
            for (int r = 0; r < 8; ++r) {
                int row = m0 + wm * 64 + i * 16 + r + half * 8;
                int col = wN + j * 16 + nc;
                long idx = (long)row * N + col;
                float v = acc[i][j][r];
                if (bias)  v += bias[col];
                if (resid) v += resid[idx];
                if (outF)  outF[idx] = v;
                if (outB)  outB[idx] = (__bf16)v;
            }
        }
    }
}

// ---------------------------------------------------------------------------
// RoPE + head reshape: qkv bf16 [T, 3H] -> qh, kh bf16 [B,NH,S,HD],
// v transposed -> vt bf16 [B,NH,HD,S] (so PV B-columns are contiguous)
// ---------------------------------------------------------------------------
__global__ __launch_bounds__(256) void rope_kernel(const __bf16* __restrict__ qkv,
                                                   __bf16* __restrict__ qh,
                                                   __bf16* __restrict__ kh,
                                                   __bf16* __restrict__ vt) {
    int g = blockIdx.x * 256 + threadIdx.x;       // over B*S*NH*HD = 4M
    int d = g & 63;
    int h = (g >> 6) & 15;
    int s = (g >> 10) & 1023;
    int b = g >> 20;
    long t = (long)b * SS + s;
    const __bf16* base = qkv + t * H3;
    int n = h * HDD + d;
    float qv = (float)base[n];
    float kv = (float)base[HH + n];
    float vv = (float)base[2 * HH + n];
    int   pr = (d < 32) ? d + 32 : d - 32;
    float qp = (float)base[h * HDD + pr];
    float kp = (float)base[HH + h * HDD + pr];
    float sg = (d < 32) ? -1.f : 1.f;
    float invf = __powf(10000.f, -(float)(d & 31) * (1.0f / 32.0f));
    float ang  = (float)s * invf;
    float cs = cosf(ang), sn = sinf(ang);
    long bh = (long)b * NHD + h;
    long rq = (bh * SS + s) * HDD + d;
    qh[rq] = (__bf16)(qv * cs + sg * qp * sn);
    kh[rq] = (__bf16)(kv * cs + sg * kp * sn);
    vt[(bh * HDD + d) * SS + s] = (__bf16)vv;
}

// ---------------------------------------------------------------------------
// Flash attention: 8 waves/block; each wave owns one 16-row Q tile of one
// (b,h). KV chunks of 32, online softmax, P routed through per-wave LDS
// tile to convert C-layout -> A-layout, PV via 4 WMMAs per chunk.
// ---------------------------------------------------------------------------
__global__ __launch_bounds__(256) void attn_kernel(const __bf16* __restrict__ qh,
                                                   const __bf16* __restrict__ kh,
                                                   const __bf16* __restrict__ vt,
                                                   __bf16* __restrict__ attnb) {
    __shared__ __bf16 pb[8][16 * 32];             // per-wave P staging

    const int lane = threadIdx.x & 31;
    const int wave = threadIdx.x >> 5;
    const int tile = blockIdx.x * 8 + wave;       // 0 .. 4095
    const int qt = tile & (SS / 16 - 1);          // 64 q-tiles per (b,h)
    const int bh = tile >> 6;
    const int q0 = qt * 16;
    const int b  = bh >> 4;
    const int h  = bh & 15;

    const __bf16* Q = qh + (long)bh * SS * HDD;
    const __bf16* K = kh + (long)bh * SS * HDD;
    const __bf16* V = vt + (long)bh * HDD * SS;

    const bf16x16 qa0 = load_a_frag_global(Q, HDD, q0, 0,  lane);
    const bf16x16 qa1 = load_a_frag_global(Q, HDD, q0, 32, lane);

    f32x8 o0 = {}, o1 = {}, o2 = {}, o3 = {};
    float mrow[8], lrow[8];
    #pragma unroll
    for (int r = 0; r < 8; ++r) { mrow[r] = -3.0e38f; lrow[r] = 0.f; }

    const int half = (lane >> 4) & 1;
    const int ncol = lane & 15;
    const int nch  = (q0 + 47) >> 5;              // causal chunk count

    for (int c = 0; c < nch; ++c) {
        const int kv0 = c * 32;
        // ---- scores: S = Q K^T  (two 16x16 tiles, K-dim = HD = 64) ----
        f32x8 s0 = {}, s1 = {};
        bf16x16 kb00 = load_b_frag(K, HDD, kv0,      0,  lane);
        bf16x16 kb01 = load_b_frag(K, HDD, kv0,      32, lane);
        bf16x16 kb10 = load_b_frag(K, HDD, kv0 + 16, 0,  lane);
        bf16x16 kb11 = load_b_frag(K, HDD, kv0 + 16, 32, lane);
        s0 = wmma_bf16(qa0, kb00, s0);
        s0 = wmma_bf16(qa1, kb01, s0);
        s1 = wmma_bf16(qa0, kb10, s1);
        s1 = wmma_bf16(qa1, kb11, s1);

        // ---- online softmax over the 16x32 chunk ----
        #pragma unroll
        for (int r = 0; r < 8; ++r) {
            int qrow = q0 + r + half * 8;
            float a  = s0[r] * 0.125f;                        // 1/sqrt(64)
            float bb = s1[r] * 0.125f;
            if (kv0 + ncol > qrow)      a  = -3.0e38f;        // causal mask
            if (kv0 + 16 + ncol > qrow) bb = -3.0e38f;
            float mx = fmaxf(a, bb);
            #pragma unroll
            for (int off = 8; off >= 1; off >>= 1)
                mx = fmaxf(mx, __shfl_xor(mx, off, 32));
            float mnew  = fmaxf(mrow[r], mx);
            float scale = __expf(mrow[r] - mnew);
            float pa = __expf(a - mnew);
            float pv = __expf(bb - mnew);
            float rs = pa + pv;
            #pragma unroll
            for (int off = 8; off >= 1; off >>= 1)
                rs += __shfl_xor(rs, off, 32);
            lrow[r] = lrow[r] * scale + rs;
            mrow[r] = mnew;
            o0[r] *= scale; o1[r] *= scale; o2[r] *= scale; o3[r] *= scale;
            // P (bf16) -> per-wave LDS tile, row-major 16x32
            pb[wave][(r + half * 8) * 32 + ncol]      = (__bf16)pa;
            pb[wave][(r + half * 8) * 32 + 16 + ncol] = (__bf16)pv;
        }
        // DS ops are in-order per wave: loads below see the stores above.
        bf16x16 pf  = load_a_frag_lds(&pb[wave][0], 32, lane);
        bf16x16 vb0 = load_b_frag(V, SS, 0,  kv0, lane);
        bf16x16 vb1 = load_b_frag(V, SS, 16, kv0, lane);
        bf16x16 vb2 = load_b_frag(V, SS, 32, kv0, lane);
        bf16x16 vb3 = load_b_frag(V, SS, 48, kv0, lane);
        o0 = wmma_bf16(pf, vb0, o0);
        o1 = wmma_bf16(pf, vb1, o1);
        o2 = wmma_bf16(pf, vb2, o2);
        o3 = wmma_bf16(pf, vb3, o3);
    }

    // ---- finalize: divide by l, reassemble heads into [T, H] bf16 ----
    #pragma unroll
    for (int r = 0; r < 8; ++r) {
        float inv = 1.0f / lrow[r];
        long t = (long)b * SS + q0 + r + half * 8;
        long o = t * HH + h * HDD;
        attnb[o + 0  + ncol] = (__bf16)(o0[r] * inv);
        attnb[o + 16 + ncol] = (__bf16)(o1[r] * inv);
        attnb[o + 32 + ncol] = (__bf16)(o2[r] * inv);
        attnb[o + 48 + ncol] = (__bf16)(o3[r] * inv);
    }
}

// ---------------------------------------------------------------------------
// Gated GELU (exact erf): pre [T, 2I] -> hid [T, I]
// ---------------------------------------------------------------------------
__global__ __launch_bounds__(256) void geglu_kernel(const __bf16* __restrict__ pre,
                                                    __bf16* __restrict__ hid) {
    long g = (long)blockIdx.x * 256 + threadIdx.x;   // over T*I
    long t = g >> 12;                                 // I = 4096
    int  i = (int)(g & (II - 1));
    float gate = (float)pre[t * I2 + i];
    float lin  = (float)pre[t * I2 + II + i];
    float ge   = 0.5f * gate * (1.0f + erff(gate * 0.70710678118f));
    hid[g] = (__bf16)(ge * lin);
}

// ---------------------------------------------------------------------------
// Host launch
// ---------------------------------------------------------------------------
extern "C" void kernel_launch(void* const* d_in, const int* in_sizes, int n_in,
                              void* d_out, int out_size, void* d_ws, size_t ws_size,
                              hipStream_t stream) {
    const float* x      = (const float*)d_in[0];
    const float* ln1_w  = (const float*)d_in[1];
    const float* ln1_b  = (const float*)d_in[2];
    const float* qkv_w  = (const float*)d_in[3];
    const float* qkv_b  = (const float*)d_in[4];
    const float* out_w  = (const float*)d_in[5];
    const float* out_b  = (const float*)d_in[6];
    const float* ln2_w  = (const float*)d_in[7];
    const float* ln2_b  = (const float*)d_in[8];
    const float* i2h_w  = (const float*)d_in[9];
    const float* i2h_b  = (const float*)d_in[10];
    const float* h2o_w  = (const float*)d_in[11];
    const float* h2o_b  = (const float*)d_in[12];
    float* out = (float*)d_out;

    // workspace carve-out
    size_t off = 0;
    auto alloc = [&](size_t bytes) -> void* {
        void* p = (char*)d_ws + off;
        off += (bytes + 255) & ~(size_t)255;
        return p;
    };
    __bf16* wq    = (__bf16*)alloc((size_t)H3 * HH * 2);       // qkv_w bf16
    __bf16* wo    = (__bf16*)alloc((size_t)HH * HH * 2);       // out_w bf16
    __bf16* wi    = (__bf16*)alloc((size_t)I2 * HH * 2);       // i2h_w bf16
    __bf16* wh    = (__bf16*)alloc((size_t)HH * II * 2);       // h2o_w bf16
    __bf16* hln   = (__bf16*)alloc((size_t)TT * HH * 2);       // LN1 out
    __bf16* qkvb  = (__bf16*)alloc((size_t)TT * H3 * 2);       // qkv proj
    __bf16* qhp   = (__bf16*)alloc((size_t)TT * HH * 2);       // q heads (rope)
    __bf16* khp   = (__bf16*)alloc((size_t)TT * HH * 2);       // k heads (rope)
    __bf16* vtp   = (__bf16*)alloc((size_t)TT * HH * 2);       // v transposed
    __bf16* attnb = (__bf16*)alloc((size_t)TT * HH * 2);       // attn heads out
    float*  attno = (float*) alloc((size_t)TT * HH * 4);       // attn_out (resid)
    __bf16* ffnln = (__bf16*)alloc((size_t)TT * HH * 2);       // LN2 out
    __bf16* pre   = (__bf16*)alloc((size_t)TT * I2 * 2);       // i2h out
    __bf16* hid   = (__bf16*)alloc((size_t)TT * II * 2);       // gated gelu
    (void)ws_size; (void)n_in; (void)in_sizes; (void)out_size;

    // 1) weights -> bf16
    {
        long n;
        n = (long)H3 * HH; cvt_bf16_kernel<<<(n + 255) / 256, 256, 0, stream>>>(qkv_w, wq, n);
        n = (long)HH * HH; cvt_bf16_kernel<<<(n + 255) / 256, 256, 0, stream>>>(out_w, wo, n);
        n = (long)I2 * HH; cvt_bf16_kernel<<<(n + 255) / 256, 256, 0, stream>>>(i2h_w, wi, n);
        n = (long)HH * II; cvt_bf16_kernel<<<(n + 255) / 256, 256, 0, stream>>>(h2o_w, wh, n);
    }
    // 2) LN1
    ln_kernel<<<TT, 128, 0, stream>>>(x, ln1_w, ln1_b, hln);
    // 3) QKV GEMM: [T,H] x [3H,H]^T -> bf16 [T,3H]
    gemm_bf16_kernel<<<dim3(H3 / 128, TT / 128), 256, 0, stream>>>(
        hln, wq, qkv_b, nullptr, nullptr, qkvb, TT, H3, HH);
    // 4) RoPE + head split + V transpose
    rope_kernel<<<(TT * HH) / 256, 256, 0, stream>>>(qkvb, qhp, khp, vtp);
    // 5) causal flash attention -> bf16 [T,H]
    attn_kernel<<<(BB * NHD * (SS / 16)) / 8, 256, 0, stream>>>(qhp, khp, vtp, attnb);
    // 6) out proj + bias + residual(x) -> f32 attn_out
    gemm_bf16_kernel<<<dim3(HH / 128, TT / 128), 256, 0, stream>>>(
        attnb, wo, out_b, x, attno, nullptr, TT, HH, HH);
    // 7) LN2
    ln_kernel<<<TT, 128, 0, stream>>>(attno, ln2_w, ln2_b, ffnln);
    // 8) i2h GEMM -> bf16 [T,2I]
    gemm_bf16_kernel<<<dim3(I2 / 128, TT / 128), 256, 0, stream>>>(
        ffnln, wi, i2h_b, nullptr, nullptr, pre, TT, I2, HH);
    // 9) gated gelu -> bf16 [T,I]
    geglu_kernel<<<((long)TT * II) / 256, 256, 0, stream>>>(pre, hid);
    // 10) h2o GEMM + bias + residual(attn_out) -> f32 d_out
    gemm_bf16_kernel<<<dim3(HH / 128, TT / 128), 256, 0, stream>>>(
        hid, wh, h2o_b, attno, out, nullptr, TT, HH, II);
}
